// GAT_Block_74715251081385
// MI455X (gfx1250) — compile-verified, matrix operand order checked by reading, not run on hardware
//
#include <hip/hip_runtime.h>
#include <math.h>

#define NND    50000
#define NED    800000
#define NGR    512
#define INF_   128
#define HEADS  8
#define HEADC  8
#define OUTF   64
#define ETOT   (NED + NND)          // edges + self loops = 850000
#define NEG_SLOPE 0.2f

#define KCHUNKS (INF_ / 32)         // 4
#define NTILES  (OUTF / 16)         // 4
#define MTILES  (NND / 16)          // 3125 (exact)

typedef __bf16 bf16x16 __attribute__((ext_vector_type(16)));
typedef float  f32x8   __attribute__((ext_vector_type(8)));

// ---------- helpers ----------
__device__ __forceinline__ float atomicMaxF(float* addr, float v) {
  // sign-split trick: works for mixed-sign values, init must be -inf
  if (v >= 0.0f)
    return __int_as_float(atomicMax((int*)addr, __float_as_int(v)));
  else
    return __uint_as_float(atomicMin((unsigned int*)addr, __float_as_uint(v)));
}

__global__ void k_fill(float* __restrict__ p, float v, int n) {
  int i = blockIdx.x * blockDim.x + threadIdx.x;
  if (i < n) p[i] = v;
}

// ---------- 0) pre-pack W into the WMMA B-operand lane layout (bf16) ----------
// Wb index = ((nt*KCHUNKS + kc)*32 + lane)*16 + i
// lane<16 holds K = {kc*32 + 0..7, kc*32 + 16..23}; lane>=16 shifts K by 8.
// col = nt*16 + lane%16
__global__ void k_packW(const float* __restrict__ W, __bf16* __restrict__ Wb) {
  int idx = blockIdx.x * blockDim.x + threadIdx.x;   // total 4*4*32*16 = 8192
  if (idx >= NTILES * KCHUNKS * 32 * 16) return;
  int i    = idx & 15;
  int lane = (idx >> 4) & 31;
  int kc   = (idx >> 9) & (KCHUNKS - 1);
  int nt   = idx >> 11;
  int col  = nt * 16 + (lane & 15);
  int koff = (lane >> 4) * 8;
  int k    = kc * 32 + ((i < 8) ? (koff + i) : (16 + koff + (i - 8)));
  Wb[idx]  = (__bf16)W[k * OUTF + col];
}

// ---------- 1) h = x @ W  via bf16 WMMA, fp32 accumulate ----------
__global__ void k_gemm_wmma(const float* __restrict__ x,
                            const __bf16* __restrict__ Wb,
                            float* __restrict__ h) {
  int wave = (blockIdx.x * blockDim.x + threadIdx.x) >> 5;   // wave-uniform
  int lane = threadIdx.x & 31;
  if (wave >= MTILES * NTILES) return;                       // whole-wave exit
  int mt = wave / NTILES, nt = wave % NTILES;
  int m0 = mt * 16, n0 = nt * 16;

  int l16  = lane & 15;
  int koff = (lane >> 4) * 8;       // 0 for lanes 0-15, 8 for lanes 16-31
  int row  = m0 + l16;              // A: M = lane%16

  f32x8 acc = {};
#pragma unroll
  for (int kc = 0; kc < KCHUNKS; ++kc) {
    int kk = kc * 32;
    // A (16x32 bf16): lane holds K = kk+koff+0..7 and kk+16+koff+0..7
    const float* xp = x + row * INF_ + kk + koff;
    bf16x16 a;
#pragma unroll
    for (int i = 0; i < 8; ++i) {
      a[i]     = (__bf16)xp[i];
      a[i + 8] = (__bf16)xp[16 + i];
    }
    // B: pre-packed, 32 bytes per lane per chunk (2x b128)
    bf16x16 b = *(const bf16x16*)(Wb + (((nt * KCHUNKS + kc) * 32) + lane) * 16);
    acc = __builtin_amdgcn_wmma_f32_16x16x32_bf16(
        /*neg_a=*/false, a, /*neg_b=*/false, b,
        /*c_mod=*/(short)0, acc, /*reuse_a=*/false, /*reuse_b=*/false);
  }
  // C/D layout: VGPR r -> row m0 + r + 8*(lane>=16), col = n0 + lane%16
  int mb = m0 + koff;
  int col = n0 + l16;
#pragma unroll
  for (int r = 0; r < 8; ++r)
    h[(mb + r) * OUTF + col] = acc[r];
}

// ---------- 2) per-(node,head) attention logits ----------
__global__ void k_att(const float* __restrict__ h,
                      const float* __restrict__ att_src,
                      const float* __restrict__ att_dst,
                      float* __restrict__ a_s, float* __restrict__ a_d) {
  int i = blockIdx.x * blockDim.x + threadIdx.x;
  if (i >= NND * HEADS) return;
  int n = i >> 3, hh = i & 7;
  const float* hp = h + n * OUTF + hh * HEADC;
  float s = 0.f, d = 0.f;
#pragma unroll
  for (int c = 0; c < HEADC; ++c) {
    s += hp[c] * att_src[hh * HEADC + c];
    d += hp[c] * att_dst[hh * HEADC + c];
  }
  a_s[i] = s; a_d[i] = d;
}

__device__ __forceinline__ void edge_sd(int e, const int* __restrict__ src,
                                        const int* __restrict__ dst,
                                        int& s, int& d) {
  if (e < NED) { s = src[e]; d = dst[e]; }
  else         { s = d = e - NED; }     // self loop
}

// ---------- 3a) segment max of leaky-relu logits over dst ----------
__global__ void k_edge_max(const int* __restrict__ src, const int* __restrict__ dst,
                           const float* __restrict__ a_s, const float* __restrict__ a_d,
                           float* __restrict__ m) {
  int i = blockIdx.x * blockDim.x + threadIdx.x;
  if (i >= ETOT * HEADS) return;
  int e = i >> 3, hh = i & 7, s, d;
  edge_sd(e, src, dst, s, d);
  float v = a_s[s * HEADS + hh] + a_d[d * HEADS + hh];
  v = (v > 0.f) ? v : NEG_SLOPE * v;
  atomicMaxF(&m[d * HEADS + hh], v);
}

// ---------- 3b) exp(e - m) per edge, denom per dst ----------
__global__ void k_edge_exp(const int* __restrict__ src, const int* __restrict__ dst,
                           const float* __restrict__ a_s, const float* __restrict__ a_d,
                           const float* __restrict__ m,
                           float* __restrict__ exb, float* __restrict__ den) {
  int i = blockIdx.x * blockDim.x + threadIdx.x;
  if (i >= ETOT * HEADS) return;
  int e = i >> 3, hh = i & 7, s, d;
  edge_sd(e, src, dst, s, d);
  float v = a_s[s * HEADS + hh] + a_d[d * HEADS + hh];
  v = (v > 0.f) ? v : NEG_SLOPE * v;
  float ex = __expf(v - m[d * HEADS + hh]);
  exb[i] = ex;
  atomicAdd(&den[d * HEADS + hh], ex);
}

// ---------- 3c) out[dst] += h[src] * alpha ----------
__global__ void k_edge_aggr(const int* __restrict__ src, const int* __restrict__ dst,
                            const float* __restrict__ h, const float* __restrict__ exb,
                            const float* __restrict__ den, float* __restrict__ out) {
  int i = blockIdx.x * blockDim.x + threadIdx.x;
  if (i >= ETOT * OUTF) return;
  int e = i >> 6, f = i & 63, hh = f >> 3, s, d;
  edge_sd(e, src, dst, s, d);
  float alpha = exb[e * HEADS + hh] / (den[d * HEADS + hh] + 1e-16f);
  atomicAdd(&out[d * OUTF + f], h[s * OUTF + f] * alpha);
}

// ---------- 4) bias ----------
__global__ void k_bias(float* __restrict__ out, const float* __restrict__ bias) {
  int i = blockIdx.x * blockDim.x + threadIdx.x;
  if (i < NND * OUTF) out[i] += bias[i & 63];
}

// ---------- 5) GraphConv aggregation over ORIGINAL edges ----------
__global__ void k_aggr2(const int* __restrict__ src, const int* __restrict__ dst,
                        const float* __restrict__ out, float* __restrict__ aggr) {
  int i = blockIdx.x * blockDim.x + threadIdx.x;
  if (i >= NED * OUTF) return;
  int e = i >> 6, f = i & 63;
  atomicAdd(&aggr[dst[e] * OUTF + f], out[src[e] * OUTF + f]);
}

// ---------- 6a) pooling score + per-graph max ----------
__global__ void k_score_max(const float* __restrict__ aggr, const float* __restrict__ out,
                            const float* __restrict__ w_rel, const float* __restrict__ b_rel,
                            const float* __restrict__ w_root, const int* __restrict__ batch,
                            float* __restrict__ score, float* __restrict__ smax) {
  int n = blockIdx.x * blockDim.x + threadIdx.x;
  if (n >= NND) return;
  float sc = b_rel[0];
#pragma unroll 8
  for (int f = 0; f < OUTF; ++f)
    sc += aggr[n * OUTF + f] * w_rel[f] + out[n * OUTF + f] * w_root[f];
  score[n] = sc;
  atomicMaxF(&smax[batch[n]], sc);
}

// ---------- 6b) exp / per-graph sum ----------
__global__ void k_score_exp(const int* __restrict__ batch, const float* __restrict__ smax,
                            float* __restrict__ score, float* __restrict__ ssum) {
  int n = blockIdx.x * blockDim.x + threadIdx.x;
  if (n >= NND) return;
  float ex = __expf(score[n] - smax[batch[n]]);
  score[n] = ex;
  atomicAdd(&ssum[batch[n]], ex);
}

// ---------- 7) scale + global_add_pool ----------
__global__ void k_pool(const float* __restrict__ out, const int* __restrict__ batch,
                       const float* __restrict__ score, const float* __restrict__ ssum,
                       float* __restrict__ g) {
  int i = blockIdx.x * blockDim.x + threadIdx.x;
  if (i >= NND * OUTF) return;
  int n = i >> 6;
  int b = batch[n];
  float sc = score[n] / (ssum[b] + 1e-16f);
  atomicAdd(&g[b * OUTF + (i & 63)], out[i] * sc);
}

// ---------- launch ----------
static inline int nb(long long n, int t) { return (int)((n + t - 1) / t); }

extern "C" void kernel_launch(void* const* d_in, const int* in_sizes, int n_in,
                              void* d_out, int out_size, void* d_ws, size_t ws_size,
                              hipStream_t stream) {
  (void)in_sizes; (void)n_in; (void)out_size; (void)ws_size;
  const float* x       = (const float*)d_in[0];
  const int*   ei      = (const int*)  d_in[1];   // [2, NED] int32 (JAX default x64-off)
  const int*   batch   = (const int*)  d_in[2];
  const float* W       = (const float*)d_in[3];
  const float* att_src = (const float*)d_in[4];
  const float* att_dst = (const float*)d_in[5];
  const float* bias    = (const float*)d_in[6];
  const float* w_rel   = (const float*)d_in[7];
  const float* b_rel   = (const float*)d_in[8];
  const float* w_root  = (const float*)d_in[9];

  const int* src = ei;
  const int* dst = ei + NED;

  float* out = (float*)d_out;                 // [NND, 64]
  float* g   = out + (long long)NND * OUTF;   // [NGR, 64]

  // workspace layout (floats)
  float* ws = (float*)d_ws;
  long long off = 0;
  float* h    = ws + off; off += (long long)NND * OUTF;   // 3.2M
  float* a_s  = ws + off; off += (long long)NND * HEADS;
  float* a_d  = ws + off; off += (long long)NND * HEADS;
  float* m    = ws + off; off += (long long)NND * HEADS;
  float* den  = ws + off; off += (long long)NND * HEADS;
  float* exb  = ws + off; off += (long long)ETOT * HEADS; // 6.8M
  float* aggr = ws + off; off += (long long)NND * OUTF;   // 3.2M
  float* score= ws + off; off += NND;
  float* smax = ws + off; off += NGR;
  float* ssum = ws + off; off += NGR;
  off = (off + 7) & ~7LL;                      // 32-byte align for bf16x16 loads
  __bf16* Wb  = (__bf16*)(ws + off);           // 8192 bf16 = 16 KB

  const int T = 256;
  const float NINF = -INFINITY;

  // init accumulators every call (graph-replay safe)
  k_fill<<<nb((long long)NND * OUTF + (long long)NGR * OUTF, T), T, 0, stream>>>(out, 0.f, NND * OUTF + NGR * OUTF);
  k_fill<<<nb(NND * HEADS, T), T, 0, stream>>>(m, NINF, NND * HEADS);
  k_fill<<<nb(NND * HEADS, T), T, 0, stream>>>(den, 0.f, NND * HEADS);
  k_fill<<<nb((long long)NND * OUTF, T), T, 0, stream>>>(aggr, 0.f, NND * OUTF);
  k_fill<<<nb(NGR, T), T, 0, stream>>>(smax, NINF, NGR);
  k_fill<<<nb(NGR, T), T, 0, stream>>>(ssum, 0.f, NGR);

  // 0) pack W into WMMA B-operand layout (16 KB, L1/L2 resident)
  k_packW<<<nb(NTILES * KCHUNKS * 32 * 16, T), T, 0, stream>>>(W, Wb);

  // 1) GEMM: 3125 * 4 waves, 8 waves per 256-thread block
  {
    int waves = MTILES * NTILES;
    k_gemm_wmma<<<nb((long long)waves * 32, T), T, 0, stream>>>(x, Wb, h);
  }
  // 2) attention logits
  k_att<<<nb(NND * HEADS, T), T, 0, stream>>>(h, att_src, att_dst, a_s, a_d);
  // 3) stable segment softmax + weighted aggregation
  k_edge_max <<<nb((long long)ETOT * HEADS, T), T, 0, stream>>>(src, dst, a_s, a_d, m);
  k_edge_exp <<<nb((long long)ETOT * HEADS, T), T, 0, stream>>>(src, dst, a_s, a_d, m, exb, den);
  k_edge_aggr<<<nb((long long)ETOT * OUTF,  T), T, 0, stream>>>(src, dst, h, exb, den, out);
  // 4) bias
  k_bias<<<nb((long long)NND * OUTF, T), T, 0, stream>>>(out, bias);
  // 5) GraphConv aggregation (original edges)
  k_aggr2<<<nb((long long)NED * OUTF, T), T, 0, stream>>>(src, dst, out, aggr);
  // 6) SAGPool score softmax per graph
  k_score_max<<<nb(NND, T), T, 0, stream>>>(aggr, out, w_rel, b_rel, w_root, batch, score, smax);
  k_score_exp<<<nb(NND, T), T, 0, stream>>>(batch, smax, score, ssum);
  // 7) scale + global add pool
  k_pool<<<nb((long long)NND * OUTF, T), T, 0, stream>>>(out, batch, score, ssum, g);
}
